// PyramidAttentionPooling_50646254355108
// MI455X (gfx1250) — matrix operand order repaired and checked by reading, not compiled, not run on hardware
//
#include <hip/hip_runtime.h>
#include <cstdint>
#include <cstddef>

#define B_   8
#define C_   512
#define KQ_  128
#define EPS_ 1e-5f

typedef __attribute__((ext_vector_type(16))) __bf16 bf16x16;
typedef __attribute__((ext_vector_type(8)))  float  floatx8;
typedef __attribute__((ext_vector_type(8)))  unsigned short ushort8;
typedef __attribute__((ext_vector_type(4)))  unsigned int   uint32x4;
typedef __attribute__((ext_vector_type(8)))  int            int32x8;
typedef __attribute__((ext_vector_type(4)))  int            int32x4;

union FragAB {
  bf16x16 v;
  ushort8 h[2];
};

#ifdef __has_builtin
#define HAS_BI(x) __has_builtin(x)
#else
#define HAS_BI(x) 0
#endif

#if HAS_BI(__builtin_amdgcn_tensor_load_to_lds)
#define HAVE_TDM 1
#else
#define HAVE_TDM 0
#endif

static __device__ __forceinline__ void wait_tensor0() {
#if defined(__HIP_DEVICE_COMPILE__)
#if HAS_BI(__builtin_amdgcn_s_wait_tensorcnt)
  __builtin_amdgcn_s_wait_tensorcnt(0);
#else
  asm volatile("s_wait_tensorcnt 0x0" ::: "memory");
#endif
#endif
}

// TDM: stage a [16 rows x 1024B] weight tile (row stride 32KB) into LDS.
// D# per CDNA5 ISA ch.8: group0 = {count=1, lds_addr, global_addr, type=2},
// group1 = {mask=0, data_size=1B, tensor_dim0/1, tile_dim0=1024B, tile_dim1=16,
//           tensor_dim0_stride=32768B}. Groups 2/3 zero (2D tensor).
static __device__ __forceinline__ void tdm_stage_wtile(const unsigned short* gsrc,
                                                       unsigned short* ldst) {
#if defined(__HIP_DEVICE_COMPILE__) && HAVE_TDM
  unsigned long long ga = (unsigned long long)(uintptr_t)gsrc;
  uint32x4 g0;
  g0[0] = 1u;                                            // count=1, user mode
  g0[1] = (unsigned int)(uintptr_t)ldst;                 // LDS byte address
  g0[2] = (unsigned int)(ga & 0xFFFFFFFFull);            // global_addr[31:0]
  g0[3] = (unsigned int)((ga >> 32) & 0x01FFFFFFull) | (2u << 30); // [56:32]|type=2
  int32x8 g1;
  g1[0] = 0;                                             // wg_mask=0, data_size=1B
  g1[1] = (int)(0x8000u << 16);                          // tensor_dim0 = 32768
  g1[2] = (int)(64u << 16);                              // tensor_dim1 = 64
  g1[3] = (int)(1024u << 16);                            // tile_dim0 = 1024 bytes
  g1[4] = 16;                                            // tile_dim1 = 16 rows
  g1[5] = 32768;                                         // dim0 stride = 32768 B
  g1[6] = 0;
  g1[7] = 0;
  int32x4 gz = {0, 0, 0, 0};
#if __clang_major__ >= 23
  int32x8 gz8 = {0, 0, 0, 0, 0, 0, 0, 0};
  __builtin_amdgcn_tensor_load_to_lds(g0, g1, gz, gz, gz8, 0);
#else
  __builtin_amdgcn_tensor_load_to_lds(g0, g1, gz, gz, 0);
#endif
#else
  (void)gsrc; (void)ldst;
#endif
}

static __device__ __forceinline__ unsigned short f32_to_bf16_rne(float f) {
  unsigned int u = __float_as_uint(f);
  unsigned int r = (u + 0x7FFFu + ((u >> 16) & 1u)) >> 16;
  return (unsigned short)r;
}

// ---------------------------------------------------------------------------
// Packing kernels
// ---------------------------------------------------------------------------

// x (B,C,H,W) f32  ->  (B,H,W,C) bf16
__global__ void k_pack_x_nhwc(const float* __restrict__ x,
                              unsigned short* __restrict__ out,
                              int H, int W) {
  size_t i = (size_t)blockIdx.x * blockDim.x + threadIdx.x;
  size_t total = (size_t)B_ * C_ * H * W;
  if (i >= total) return;
  int c = (int)(i % C_);
  size_t t = i / C_;
  int w = (int)(t % W); t /= W;
  int h = (int)(t % H);
  int b = (int)(t / H);
  float v = x[(((size_t)b * C_ + c) * H + h) * W + w];
  out[i] = f32_to_bf16_rne(v);
}

// conv weights (O,I,3,3) f32 -> packed [(4608/32) kchunks][O][32] bf16
// k ordering: k = (ky*3+kx)*C + c  (32-chunks never straddle the 512-c block)
__global__ void k_pack_conv_w(const float* __restrict__ w,
                              unsigned short* __restrict__ wp) {
  size_t i = (size_t)blockIdx.x * blockDim.x + threadIdx.x;
  size_t total = (size_t)C_ * 9 * C_;
  if (i >= total) return;
  int j = (int)(i % 32);
  size_t t2 = i / 32;
  int o = (int)(t2 % C_);
  int t = (int)(t2 / C_);
  int k = t * 32 + j;
  int kyx = k / C_;
  int c   = k % C_;
  int ky = kyx / 3, kx = kyx % 3;
  float v = w[(((size_t)o * C_ + c) * 3 + ky) * 3 + kx];
  wp[i] = f32_to_bf16_rne(v);
}

// 1x1 weights (N,C) f32 -> packed [(C/32) kchunks][N][32] bf16
__global__ void k_pack_w1x1(const float* __restrict__ w,
                            unsigned short* __restrict__ wp, int N) {
  size_t i = (size_t)blockIdx.x * blockDim.x + threadIdx.x;
  size_t total = (size_t)N * C_;
  if (i >= total) return;
  int j = (int)(i % 32);
  size_t t2 = i / 32;
  int n = (int)(t2 % N);
  int t = (int)(t2 / N);
  wp[i] = f32_to_bf16_rne(w[(size_t)n * C_ + t * 32 + j]);
}

__global__ void k_fill_u16(unsigned short* __restrict__ p, size_t n) {
  size_t i = (size_t)blockIdx.x * blockDim.x + threadIdx.x;
  if (i < n) p[i] = 0;
}

// ---------------------------------------------------------------------------
// Implicit-GEMM 3x3 stride-2 conv, bf16 WMMA, fp32 accumulate.
// Grid: (Mtiles/4, C/16), block 128 (4 waves). All 4 waves share one N tile,
// whose weights are TDM-staged into double-buffered LDS (16KB per buffer):
// wave0 issues tensor_load_to_lds for tap kyx+1 while everyone computes tap
// kyx from LDS, then s_wait_tensorcnt + workgroup barrier. The chunk loop has
// a wave-uniform __all(valid) fast path (interior tiles: unconditional A
// loads, no EXEC juggling / zero-fill) and a predicated boundary path.
// ---------------------------------------------------------------------------
__global__ void k_conv3s2_wmma(const unsigned short* __restrict__ xin, // NHWC bf16
                               const unsigned short* __restrict__ wp,  // packed
                               float* __restrict__ y,                  // (M,C) f32
                               int IH, int IW, int OH, int OW) {
  __shared__ unsigned short bstage[2][16 * 16 * 32];  // 2 x 16KB
  const int wid  = threadIdx.x >> 5;
  const int lane = threadIdx.x & 31;
  const int tn = blockIdx.y;
  const int g  = lane >> 4;
  const int ln = lane & 15;
  const int tm = blockIdx.x * 4 + wid;

  const int m   = tm * 16 + ln;             // A-fragment row (output position)
  const int ow_ = m % OW;
  int tq = m / OW;
  const int oh_ = tq % OH;
  const int b   = tq / OH;
  const int col = tn * 16 + ln;             // B-fragment column (out channel)

  // weight source for this block's N tile: rows t (kchunks), 1KB each,
  // row stride C_*32 ushorts (32KB)
  const unsigned short* wsrc = wp + ((size_t)tn * 16) * 32;

  floatx8 acc = {0.f, 0.f, 0.f, 0.f, 0.f, 0.f, 0.f, 0.f};
  const ushort8 zero8 = {0, 0, 0, 0, 0, 0, 0, 0};

#if HAVE_TDM
  if (wid == 0) {
    tdm_stage_wtile(wsrc, &bstage[0][0]);
    wait_tensor0();
  }
  __syncthreads();
#else
  for (int i = threadIdx.x; i < 16 * 64; i += 128) {
    int r = i >> 6, j = i & 63;
    ((ushort8*)&bstage[0][0])[r * 64 + j] =
        *((const ushort8*)(wsrc + (size_t)r * C_ * 32) + j);
  }
  __syncthreads();
#endif

  for (int kyx = 0; kyx < 9; ++kyx) {
    const int p = kyx & 1;
#if HAVE_TDM
    if (kyx + 1 < 9 && wid == 0)
      tdm_stage_wtile(wsrc + (size_t)((kyx + 1) * 16) * C_ * 32,
                      &bstage[1 - p][0]);
#else
    if (kyx + 1 < 9) {
      for (int i = threadIdx.x; i < 16 * 64; i += 128) {
        int r = i >> 6, j = i & 63;
        ((ushort8*)&bstage[1 - p][0])[r * 64 + j] =
            *((const ushort8*)(wsrc + (size_t)((kyx + 1) * 16 + r) * C_ * 32) + j);
      }
    }
#endif
    // hoisted per-tap addressing
    const int ky = kyx / 3, kx = kyx - ky * 3;
    const int iy = oh_ * 2 - 1 + ky;
    const int ix = ow_ * 2 - 1 + kx;
    const bool vld = ((unsigned)iy < (unsigned)IH) && ((unsigned)ix < (unsigned)IW);
    const unsigned short* ap0 = xin + (((size_t)b * IH + iy) * IW + ix) * C_;

    if (__all(vld ? 1 : 0)) {
      // interior fast path: whole wave in-bounds, unconditional loads
#pragma unroll 4
      for (int t16 = 0; t16 < 16; ++t16) {
        FragAB A;
        const unsigned short* ap = ap0 + t16 * 32;
        A.h[0] = *(const ushort8*)(ap + g * 16);       // K = g*16 .. g*16+7
        A.h[1] = *(const ushort8*)(ap + g * 16 + 8);   // K = g*16+8 ..
        FragAB Bf;
        const unsigned short* bp = &bstage[p][(t16 * 16 + ln) * 32 + g * 16];
        Bf.h[0] = *(const ushort8*)(bp);
        Bf.h[1] = *(const ushort8*)(bp + 8);
        acc = __builtin_amdgcn_wmma_f32_16x16x32_bf16(
            false, A.v, false, Bf.v, (short)0, acc, false, false);
      }
    } else {
      // boundary path: per-lane zero padding
#pragma unroll 4
      for (int t16 = 0; t16 < 16; ++t16) {
        FragAB A;
        if (vld) {
          const unsigned short* ap = ap0 + t16 * 32;
          A.h[0] = *(const ushort8*)(ap + g * 16);
          A.h[1] = *(const ushort8*)(ap + g * 16 + 8);
        } else {
          A.h[0] = zero8; A.h[1] = zero8;
        }
        FragAB Bf;
        const unsigned short* bp = &bstage[p][(t16 * 16 + ln) * 32 + g * 16];
        Bf.h[0] = *(const ushort8*)(bp);
        Bf.h[1] = *(const ushort8*)(bp + 8);
        acc = __builtin_amdgcn_wmma_f32_16x16x32_bf16(
            false, A.v, false, Bf.v, (short)0, acc, false, false);
      }
    }
#if HAVE_TDM
    if (wid == 0) wait_tensor0();
#endif
    __syncthreads();
  }

  const int rowbase = tm * 16 + g * 8;
#pragma unroll
  for (int r = 0; r < 8; ++r)
    y[(size_t)(rowbase + r) * C_ + col] = acc[r];
}

// ---------------------------------------------------------------------------
// BatchNorm: per-channel mean/var over M rows, then apply + ReLU -> bf16 NHWC
// ---------------------------------------------------------------------------
__global__ void k_bn_stats(const float* __restrict__ y,
                           float* __restrict__ stats, int M) {
  const int c = blockIdx.x;
  const int tid = threadIdx.x;
  float s = 0.f, s2 = 0.f;
  for (int m = tid; m < M; m += 256) {
    float v = y[(size_t)m * C_ + c];
    s += v; s2 += v * v;
  }
  __shared__ float r1[256], r2[256];
  r1[tid] = s; r2[tid] = s2;
  __syncthreads();
  for (int off2 = 128; off2 > 0; off2 >>= 1) {
    if (tid < off2) { r1[tid] += r1[tid + off2]; r2[tid] += r2[tid + off2]; }
    __syncthreads();
  }
  if (tid == 0) {
    float mean = r1[0] / (float)M;
    float var  = r2[0] / (float)M - mean * mean;
    stats[c] = mean;
    stats[C_ + c] = var;
  }
}

__global__ void k_bn_apply(const float* __restrict__ y,
                           const float* __restrict__ stats,
                           const float* __restrict__ gw,
                           const float* __restrict__ gb,
                           unsigned short* __restrict__ out, size_t total) {
  size_t i = (size_t)blockIdx.x * blockDim.x + threadIdx.x;
  if (i >= total) return;
  int c = (int)(i % C_);
  float v = (y[i] - stats[c]) * rsqrtf(stats[C_ + c] + EPS_) * gw[c] + gb[c];
  out[i] = f32_to_bf16_rne(fmaxf(v, 0.f));
}

// ---------------------------------------------------------------------------
// 1x1 projection GEMM (q/k/v), 16x32 tile per wave (A reused by 2 WMMAs).
// mode 0: bf16 (M,N). mode 1 (v): bf16 transposed (b,C,Npad) + fp32 (b,C,HW).
// ---------------------------------------------------------------------------
__global__ void k_gemm1x1_wmma(const unsigned short* __restrict__ A_,
                               const unsigned short* __restrict__ wp,
                               const float* __restrict__ bias,
                               int Ncols, int Mtiles, int HW, int Npad, int mode,
                               unsigned short* __restrict__ outQ,
                               unsigned short* __restrict__ outVb,
                               float* __restrict__ outVf) {
  const int wid  = threadIdx.x >> 5;
  const int lane = threadIdx.x & 31;
  const int tm = blockIdx.x * 4 + wid;
  if (tm >= Mtiles) return;
  const int g  = lane >> 4;
  const int ln = lane & 15;
  const int row  = tm * 16 + ln;
  const int col0 = blockIdx.y * 32 + ln;
  const int col1 = col0 + 16;

  floatx8 acc0 = {0.f, 0.f, 0.f, 0.f, 0.f, 0.f, 0.f, 0.f};
  floatx8 acc1 = {0.f, 0.f, 0.f, 0.f, 0.f, 0.f, 0.f, 0.f};
  for (int t = 0; t < C_ / 32; ++t) {
    FragAB A;
    const unsigned short* ap = A_ + (size_t)row * C_ + t * 32;
    A.h[0] = *(const ushort8*)(ap + g * 16);
    A.h[1] = *(const ushort8*)(ap + g * 16 + 8);
    FragAB B0, B1;
    const unsigned short* bp0 = wp + ((size_t)t * Ncols + col0) * 32 + g * 16;
    const unsigned short* bp1 = wp + ((size_t)t * Ncols + col1) * 32 + g * 16;
    B0.h[0] = *(const ushort8*)(bp0);
    B0.h[1] = *(const ushort8*)(bp0 + 8);
    B1.h[0] = *(const ushort8*)(bp1);
    B1.h[1] = *(const ushort8*)(bp1 + 8);
    __builtin_prefetch(bp0 + (size_t)Ncols * 32, 0, 1);
    acc0 = __builtin_amdgcn_wmma_f32_16x16x32_bf16(
        false, A.v, false, B0.v, (short)0, acc0, false, false);
    acc1 = __builtin_amdgcn_wmma_f32_16x16x32_bf16(
        false, A.v, false, B1.v, (short)0, acc1, false, false);
  }

  const float b0 = bias[col0];
  const float b1 = bias[col1];
#pragma unroll
  for (int r = 0; r < 8; ++r) {
    int mrow = tm * 16 + g * 8 + r;
    float v0 = acc0[r] + b0;
    float v1 = acc1[r] + b1;
    if (mode == 0) {
      outQ[(size_t)mrow * Ncols + col0] = f32_to_bf16_rne(v0);
      outQ[(size_t)mrow * Ncols + col1] = f32_to_bf16_rne(v1);
    } else {
      int b = mrow / HW, hw = mrow % HW;
      outVb[((size_t)b * C_ + col0) * Npad + hw] = f32_to_bf16_rne(v0);
      outVb[((size_t)b * C_ + col1) * Npad + hw] = f32_to_bf16_rne(v1);
      outVf[((size_t)b * C_ + col0) * HW + hw]  = v0;
      outVf[((size_t)b * C_ + col1) * HW + hw]  = v1;
    }
  }
}

// ---------------------------------------------------------------------------
// energy[b,n,m] = sum_kc q[b,n,kc]*k[b,m,kc]; 16x32 tile per wave, uniform
// guard on the second column tile (level-4 has 9 column tiles).
// ---------------------------------------------------------------------------
__global__ void k_energy_wmma(const unsigned short* __restrict__ q,
                              const unsigned short* __restrict__ kk,
                              float* __restrict__ e, int HW) {
  const int wid  = threadIdx.x >> 5;
  const int lane = threadIdx.x & 31;
  const int Stiles = HW / 16;
  const int tm = blockIdx.x * 4 + wid;
  if (tm >= Stiles) return;
  const int b  = blockIdx.z;
  const int g  = lane >> 4;
  const int ln = lane & 15;
  const unsigned short* qb = q  + (size_t)b * HW * KQ_;
  const unsigned short* kb = kk + (size_t)b * HW * KQ_;
  const int row  = tm * 16 + ln;
  const int col0 = blockIdx.y * 32 + ln;
  const int col1 = col0 + 16;
  const bool has1 = (int)(blockIdx.y * 2 + 1) < Stiles;   // wave-uniform

  floatx8 acc0 = {0.f, 0.f, 0.f, 0.f, 0.f, 0.f, 0.f, 0.f};
  floatx8 acc1 = {0.f, 0.f, 0.f, 0.f, 0.f, 0.f, 0.f, 0.f};
  for (int t = 0; t < KQ_ / 32; ++t) {
    FragAB A;
    const unsigned short* ap = qb + (size_t)row * KQ_ + t * 32;
    A.h[0] = *(const ushort8*)(ap + g * 16);
    A.h[1] = *(const ushort8*)(ap + g * 16 + 8);
    FragAB B0;
    const unsigned short* bp0 = kb + (size_t)col0 * KQ_ + t * 32 + g * 16;
    B0.h[0] = *(const ushort8*)(bp0);
    B0.h[1] = *(const ushort8*)(bp0 + 8);
    acc0 = __builtin_amdgcn_wmma_f32_16x16x32_bf16(
        false, A.v, false, B0.v, (short)0, acc0, false, false);
    if (has1) {
      FragAB B1;
      const unsigned short* bp1 = kb + (size_t)col1 * KQ_ + t * 32 + g * 16;
      B1.h[0] = *(const ushort8*)(bp1);
      B1.h[1] = *(const ushort8*)(bp1 + 8);
      acc1 = __builtin_amdgcn_wmma_f32_16x16x32_bf16(
          false, A.v, false, B1.v, (short)0, acc1, false, false);
    }
  }
  float* eb = e + (size_t)b * HW * HW;
#pragma unroll
  for (int r = 0; r < 8; ++r) {
    eb[(size_t)(tm * 16 + g * 8 + r) * HW + col0] = acc0[r];
    if (has1)
      eb[(size_t)(tm * 16 + g * 8 + r) * HW + col1] = acc1[r];
  }
}

// ---------------------------------------------------------------------------
// Row softmax (fp32) -> attn bf16, rows padded with zeros to Npad
// ---------------------------------------------------------------------------
__global__ void k_softmax(const float* __restrict__ e,
                          unsigned short* __restrict__ a, int HW, int Npad) {
  const int row = blockIdx.x;                 // b*HW + n
  const float* er = e + (size_t)row * HW;
  unsigned short* ar = a + (size_t)row * Npad;
  __shared__ float red[256];
  const int tid = threadIdx.x;

  float mx = -3.0e38f;
  for (int m = tid; m < HW; m += 256) mx = fmaxf(mx, er[m]);
  red[tid] = mx; __syncthreads();
  for (int off2 = 128; off2 > 0; off2 >>= 1) {
    if (tid < off2) red[tid] = fmaxf(red[tid], red[tid + off2]);
    __syncthreads();
  }
  mx = red[0];
  __syncthreads();

  float s = 0.f;
  for (int m = tid; m < HW; m += 256) s += __expf(er[m] - mx);
  red[tid] = s; __syncthreads();
  for (int off2 = 128; off2 > 0; off2 >>= 1) {
    if (tid < off2) red[tid] += red[tid + off2];
    __syncthreads();
  }
  const float inv = 1.0f / red[0];
  for (int m = tid; m < Npad; m += 256)
    ar[m] = (m < HW) ? f32_to_bf16_rne(__expf(er[m] - mx) * inv)
                     : (unsigned short)0;
}

// ---------------------------------------------------------------------------
// p[b,c,n] = g * sum_m attn[b,n,m]*v[b,c,m] + v[b,c,n]; 16x32 tile per wave.
// ---------------------------------------------------------------------------
__global__ void k_attnv_wmma(const unsigned short* __restrict__ attn,
                             const unsigned short* __restrict__ vbf,
                             const float* __restrict__ vf,
                             const float* __restrict__ gptr,
                             float* __restrict__ p, int HW, int Npad) {
  const int wid  = threadIdx.x >> 5;
  const int lane = threadIdx.x & 31;
  const int Stiles = HW / 16;
  const int tm = blockIdx.x * 4 + wid;
  if (tm >= Stiles) return;
  const int b  = blockIdx.z;
  const int g  = lane >> 4;
  const int ln = lane & 15;
  const unsigned short* ab = attn + (size_t)b * HW * Npad;
  const unsigned short* vb = vbf  + (size_t)b * C_ * Npad;
  const int row  = tm * 16 + ln;                 // n
  const int col0 = blockIdx.y * 32 + ln;         // c (C/32 pairs, always even)
  const int col1 = col0 + 16;

  floatx8 acc0 = {0.f, 0.f, 0.f, 0.f, 0.f, 0.f, 0.f, 0.f};
  floatx8 acc1 = {0.f, 0.f, 0.f, 0.f, 0.f, 0.f, 0.f, 0.f};
  for (int t = 0; t < Npad / 32; ++t) {
    FragAB A;
    const unsigned short* ap = ab + (size_t)row * Npad + t * 32;
    A.h[0] = *(const ushort8*)(ap + g * 16);
    A.h[1] = *(const ushort8*)(ap + g * 16 + 8);
    FragAB B0, B1;
    const unsigned short* bp0 = vb + (size_t)col0 * Npad + t * 32 + g * 16;
    const unsigned short* bp1 = vb + (size_t)col1 * Npad + t * 32 + g * 16;
    B0.h[0] = *(const ushort8*)(bp0);
    B0.h[1] = *(const ushort8*)(bp0 + 8);
    B1.h[0] = *(const ushort8*)(bp1);
    B1.h[1] = *(const ushort8*)(bp1 + 8);
    acc0 = __builtin_amdgcn_wmma_f32_16x16x32_bf16(
        false, A.v, false, B0.v, (short)0, acc0, false, false);
    acc1 = __builtin_amdgcn_wmma_f32_16x16x32_bf16(
        false, A.v, false, B1.v, (short)0, acc1, false, false);
  }
  const float gg = gptr[0];
#pragma unroll
  for (int r = 0; r < 8; ++r) {
    int n = tm * 16 + g * 8 + r;
    size_t i0 = ((size_t)b * C_ + col0) * HW + n;
    size_t i1 = ((size_t)b * C_ + col1) * HW + n;
    p[i0] = gg * acc0[r] + vf[i0];
    p[i1] = gg * acc1[r] + vf[i1];
  }
}

// ---------------------------------------------------------------------------
// Bilinear (align-corners linspace) upsample; dst += up(src) / final combine
// ---------------------------------------------------------------------------
__global__ void k_upsample_add(float* __restrict__ dst,
                               const float* __restrict__ src,
                               int IH, int IW, int OH, int OW) {
  size_t i = (size_t)blockIdx.x * blockDim.x + threadIdx.x;
  size_t total = (size_t)B_ * C_ * OH * OW;
  if (i >= total) return;
  int ox = (int)(i % OW);
  size_t t = i / OW;
  int oy = (int)(t % OH);
  size_t bc = t / OH;
  float sy = (float)oy * (float)(IH - 1) / (float)(OH - 1);
  float sx = (float)ox * (float)(IW - 1) / (float)(OW - 1);
  int y0 = (int)sy, x0 = (int)sx;
  int y1 = min(y0 + 1, IH - 1), x1 = min(x0 + 1, IW - 1);
  float ty = sy - (float)y0, tx = sx - (float)x0;
  const float* s = src + bc * (size_t)(IH * IW);
  float v00 = s[(size_t)y0 * IW + x0], v01 = s[(size_t)y0 * IW + x1];
  float v10 = s[(size_t)y1 * IW + x0], v11 = s[(size_t)y1 * IW + x1];
  dst[i] += v00 * (1.f - ty) * (1.f - tx) + v01 * (1.f - ty) * tx +
            v10 * ty * (1.f - tx) + v11 * ty * tx;
}

__global__ void k_final_out(float* __restrict__ out, const float* __restrict__ x,
                            const float* __restrict__ p2,
                            const float* __restrict__ gamma,
                            int IH, int IW, int OH, int OW) {
  size_t i = (size_t)blockIdx.x * blockDim.x + threadIdx.x;
  size_t total = (size_t)B_ * C_ * OH * OW;
  if (i >= total) return;
  int ox = (int)(i % OW);
  size_t t = i / OW;
  int oy = (int)(t % OH);
  size_t bc = t / OH;
  float sy = (float)oy * (float)(IH - 1) / (float)(OH - 1);
  float sx = (float)ox * (float)(IW - 1) / (float)(OW - 1);
  int y0 = (int)sy, x0 = (int)sx;
  int y1 = min(y0 + 1, IH - 1), x1 = min(x0 + 1, IW - 1);
  float ty = sy - (float)y0, tx = sx - (float)x0;
  const float* s = p2 + bc * (size_t)(IH * IW);
  float v00 = s[(size_t)y0 * IW + x0], v01 = s[(size_t)y0 * IW + x1];
  float v10 = s[(size_t)y1 * IW + x0], v11 = s[(size_t)y1 * IW + x1];
  float u = v00 * (1.f - ty) * (1.f - tx) + v01 * (1.f - ty) * tx +
            v10 * ty * (1.f - tx) + v11 * ty * tx;
  out[i] = x[i] + gamma[0] * u;
}

// ---------------------------------------------------------------------------
// Host orchestration
// ---------------------------------------------------------------------------
extern "C" void kernel_launch(void* const* d_in, const int* in_sizes, int n_in,
                              void* d_out, int out_size, void* d_ws, size_t ws_size,
                              hipStream_t stream) {
  (void)in_sizes; (void)n_in; (void)out_size; (void)ws_size;

  const float* x     = (const float*)d_in[0];
  const float* gamma = (const float*)d_in[31];

  char* ws = (char*)d_ws;
  size_t off = 0;
  auto alloc = [&](size_t bytes) -> char* {
    char* p = ws + off;
    off += (bytes + 255) & ~(size_t)255;
    return p;
  };

  unsigned short* xnhwc = (unsigned short*)alloc((size_t)B_ * 96 * 96 * C_ * 2);
  unsigned short* wpc   = (unsigned short*)alloc((size_t)C_ * 9 * C_ * 2);
  float*          ybuf  = (float*)alloc((size_t)B_ * 48 * 48 * C_ * 4);
  float*          stats = (float*)alloc((size_t)2 * C_ * 4);
  unsigned short* dn[3];
  dn[0] = (unsigned short*)alloc((size_t)B_ * 2304 * C_ * 2);
  dn[1] = (unsigned short*)alloc((size_t)B_ * 576 * C_ * 2);
  dn[2] = (unsigned short*)alloc((size_t)B_ * 144 * C_ * 2);
  unsigned short* wq   = (unsigned short*)alloc((size_t)KQ_ * C_ * 2);
  unsigned short* wk   = (unsigned short*)alloc((size_t)KQ_ * C_ * 2);
  unsigned short* wv   = (unsigned short*)alloc((size_t)C_ * C_ * 2);
  unsigned short* qbuf = (unsigned short*)alloc((size_t)B_ * 2304 * KQ_ * 2);
  unsigned short* kbuf = (unsigned short*)alloc((size_t)B_ * 2304 * KQ_ * 2);
  unsigned short* vbf  = (unsigned short*)alloc((size_t)B_ * C_ * 2304 * 2);
  float*          vf32 = (float*)alloc((size_t)B_ * C_ * 2304 * 4);
  float*        energy = (float*)alloc((size_t)B_ * 2304 * 2304 * 4);
  unsigned short* attn = (unsigned short*)alloc((size_t)B_ * 2304 * 2304 * 2);
  float* pbuf[3];
  pbuf[0] = (float*)alloc((size_t)B_ * C_ * 2304 * 4);
  pbuf[1] = (float*)alloc((size_t)B_ * C_ * 576 * 4);
  pbuf[2] = (float*)alloc((size_t)B_ * C_ * 144 * 4);

  const int IHs[3] = {96, 48, 24};
  const int OHs[3] = {48, 24, 12};

  {
    size_t total = (size_t)B_ * C_ * 96 * 96;
    k_pack_x_nhwc<<<dim3((unsigned)((total + 255) / 256)), 256, 0, stream>>>(
        x, xnhwc, 96, 96);
  }

  const unsigned short* cur_in = xnhwc;
  for (int l = 0; l < 3; ++l) {
    const int base = 1 + 10 * l;
    const float* w   = (const float*)d_in[base + 0];
    const float* bnw = (const float*)d_in[base + 1];
    const float* bnb = (const float*)d_in[base + 2];
    const float* qw  = (const float*)d_in[base + 3];
    const float* qb  = (const float*)d_in[base + 4];
    const float* kw  = (const float*)d_in[base + 5];
    const float* kb  = (const float*)d_in[base + 6];
    const float* vw  = (const float*)d_in[base + 7];
    const float* vb  = (const float*)d_in[base + 8];
    const float* gl  = (const float*)d_in[base + 9];

    const int IH = IHs[l], OH = OHs[l];
    const int HW = OH * OH;
    const int M  = B_ * HW;
    const int Npad = (HW + 31) & ~31;
    const int Mtiles = M / 16;            // 1152 / 288 / 72, all % 4 == 0

    { size_t total = (size_t)C_ * 9 * C_;
      k_pack_conv_w<<<dim3((unsigned)((total + 255) / 256)), 256, 0, stream>>>(w, wpc); }
    k_conv3s2_wmma<<<dim3((unsigned)(Mtiles / 4), C_ / 16), 128, 0, stream>>>(
        cur_in, wpc, ybuf, IH, IH, OH, OH);
    k_bn_stats<<<dim3(C_), 256, 0, stream>>>(ybuf, stats, M);
    { size_t total = (size_t)M * C_;
      k_bn_apply<<<dim3((unsigned)((total + 255) / 256)), 256, 0, stream>>>(
          ybuf, stats, bnw, bnb, dn[l], total); }

    { size_t t1 = (size_t)KQ_ * C_;
      k_pack_w1x1<<<dim3((unsigned)((t1 + 255) / 256)), 256, 0, stream>>>(qw, wq, KQ_);
      k_pack_w1x1<<<dim3((unsigned)((t1 + 255) / 256)), 256, 0, stream>>>(kw, wk, KQ_);
      size_t t2 = (size_t)C_ * C_;
      k_pack_w1x1<<<dim3((unsigned)((t2 + 255) / 256)), 256, 0, stream>>>(vw, wv, C_); }

    k_gemm1x1_wmma<<<dim3((unsigned)(Mtiles / 4), KQ_ / 32), 128, 0, stream>>>(
        dn[l], wq, qb, KQ_, Mtiles, HW, Npad, 0, qbuf, nullptr, nullptr);
    k_gemm1x1_wmma<<<dim3((unsigned)(Mtiles / 4), KQ_ / 32), 128, 0, stream>>>(
        dn[l], wk, kb, KQ_, Mtiles, HW, Npad, 0, kbuf, nullptr, nullptr);
    { size_t t = (size_t)B_ * C_ * Npad;
      k_fill_u16<<<dim3((unsigned)((t + 255) / 256)), 256, 0, stream>>>(vbf, t); }
    k_gemm1x1_wmma<<<dim3((unsigned)(Mtiles / 4), C_ / 32), 128, 0, stream>>>(
        dn[l], wv, vb, C_, Mtiles, HW, Npad, 1, nullptr, vbf, vf32);

    const int Stiles = HW / 16;
    k_energy_wmma<<<dim3((unsigned)((Stiles + 3) / 4), (unsigned)((Stiles + 1) / 2), B_),
                    128, 0, stream>>>(qbuf, kbuf, energy, HW);
    k_softmax<<<dim3((unsigned)(B_ * HW)), 256, 0, stream>>>(energy, attn, HW, Npad);
    k_attnv_wmma<<<dim3((unsigned)((Stiles + 3) / 4), C_ / 32, B_), 128, 0, stream>>>(
        attn, vbf, vf32, gl, pbuf[l], HW, Npad);

    cur_in = dn[l];
  }

  { size_t t = (size_t)B_ * C_ * 24 * 24;
    k_upsample_add<<<dim3((unsigned)((t + 255) / 256)), 256, 0, stream>>>(
        pbuf[1], pbuf[2], 12, 12, 24, 24); }
  { size_t t = (size_t)B_ * C_ * 48 * 48;
    k_upsample_add<<<dim3((unsigned)((t + 255) / 256)), 256, 0, stream>>>(
        pbuf[0], pbuf[1], 24, 24, 48, 48); }
  { size_t t = (size_t)B_ * C_ * 96 * 96;
    k_final_out<<<dim3((unsigned)((t + 255) / 256)), 256, 0, stream>>>(
        (float*)d_out, x, pbuf[0], gamma, 48, 48, 96, 96); }
}
